// MASTER_17119739641935
// MI455X (gfx1250) — matrix operand order repaired and checked by reading, not compiled, not run on hardware
//
#include <hip/hip_runtime.h>
#include <math.h>

typedef __attribute__((ext_vector_type(16))) _Float16 v16h;
typedef __attribute__((ext_vector_type(8)))  _Float16 v8h;
typedef __attribute__((ext_vector_type(4)))  _Float16 v4h;
typedef __attribute__((ext_vector_type(8)))  float    v8f;

union AF { v16h v; _Float16 h[16]; };
union CF { v8f  v; float   f[8];  };

#define WMMA_F32_F16(A, B, C) \
  __builtin_amdgcn_wmma_f32_16x16x32_f16(false, (A), false, (B), (short)0, (C), false, false)

// ---------------------------------------------------------------------------
// WMMA GEMM:  C[M x 256] = A[M x Kpad](f32, lda) @ Bp(f16, fragment-packed)
// Bp layout: Bp[((k>>5)*256 + n)*32 + (k&31)]  (B-fragments contiguous)
// Block: 256 threads (8 waves 4x2), tile 128(M) x 64(N), K chunk 32.
// Requires M % 128 == 0.
// ---------------------------------------------------------------------------
__global__ __launch_bounds__(256) void k_gemm(
    const float* __restrict__ A, int lda,
    const _Float16* __restrict__ Bp, int Kpad,
    float* __restrict__ C, int M,
    const float* __restrict__ bias,
    const float* __restrict__ resid,
    int relu)
{
  __shared__ __align__(16) _Float16 sA[128][32];
  const int tid  = threadIdx.x;
  const int lane = tid & 31;
  const int lq   = lane & 15;
  const int hi   = lane >> 4;
  const int wave = tid >> 5;
  const int wm   = wave >> 1;      // 0..3 -> 32 rows each
  const int wn   = wave & 1;       // 0..1 -> 32 cols each
  const int m0   = blockIdx.y * 128;
  const int n0   = blockIdx.x * 64;

  CF acc00, acc01, acc10, acc11;
  #pragma unroll
  for (int r = 0; r < 8; ++r) {
    acc00.f[r] = 0.f; acc01.f[r] = 0.f; acc10.f[r] = 0.f; acc11.f[r] = 0.f;
  }

  const int rowS = tid >> 1;             // 0..127
  const int segS = (tid & 1) * 16;       // 0 or 16
  const int nB   = n0 + wn * 32 + lq;

  for (int k0 = 0; k0 < Kpad; k0 += 32) {
    // B fragments straight from global packed weights (2x b128 each)
    const _Float16* bp0 = Bp + ((size_t)(k0 >> 5) * 256 + nB) * 32 + hi * 16;
    const _Float16* bp1 = bp0 + 16 * 32;
    AF b0, b1;
    *(v8h*)&b0.h[0] = *(const v8h*)(bp0);
    *(v8h*)&b0.h[8] = *(const v8h*)(bp0 + 8);
    *(v8h*)&b1.h[0] = *(const v8h*)(bp1);
    *(v8h*)&b1.h[8] = *(const v8h*)(bp1 + 8);

    // Stage A tile 128x32 f32 -> f16 LDS (float4 loads, b128 stores)
    {
      const float* ap = A + (size_t)(m0 + rowS) * lda + k0 + segS;
      float4 f0 = *(const float4*)(ap);
      float4 f1 = *(const float4*)(ap + 4);
      float4 f2 = *(const float4*)(ap + 8);
      float4 f3 = *(const float4*)(ap + 12);
      v8h h0 = { (_Float16)f0.x, (_Float16)f0.y, (_Float16)f0.z, (_Float16)f0.w,
                 (_Float16)f1.x, (_Float16)f1.y, (_Float16)f1.z, (_Float16)f1.w };
      v8h h1 = { (_Float16)f2.x, (_Float16)f2.y, (_Float16)f2.z, (_Float16)f2.w,
                 (_Float16)f3.x, (_Float16)f3.y, (_Float16)f3.z, (_Float16)f3.w };
      *(v8h*)&sA[rowS][segS]     = h0;
      *(v8h*)&sA[rowS][segS + 8] = h1;
    }
    __syncthreads();

    // A fragments (2x b128 each)
    AF a0, a1;
    {
      int mA = wm * 32 + lq;
      int kb = hi * 16;
      *(v8h*)&a0.h[0] = *(const v8h*)&sA[mA][kb];
      *(v8h*)&a0.h[8] = *(const v8h*)&sA[mA][kb + 8];
      *(v8h*)&a1.h[0] = *(const v8h*)&sA[mA + 16][kb];
      *(v8h*)&a1.h[8] = *(const v8h*)&sA[mA + 16][kb + 8];
    }
    acc00.v = WMMA_F32_F16(a0.v, b0.v, acc00.v);
    acc01.v = WMMA_F32_F16(a0.v, b1.v, acc01.v);
    acc10.v = WMMA_F32_F16(a1.v, b0.v, acc10.v);
    acc11.v = WMMA_F32_F16(a1.v, b1.v, acc11.v);
    __syncthreads();
  }

  // epilogue
  const int mb = (hi << 3);
  #pragma unroll
  for (int af = 0; af < 2; ++af) {
    CF* accA = af ? &acc10 : &acc00;
    CF* accB = af ? &acc11 : &acc01;
    #pragma unroll
    for (int r = 0; r < 8; ++r) {
      int gm = m0 + wm * 32 + af * 16 + mb + r;
      int nc = n0 + wn * 32 + lq;
      float v0 = accA->f[r], v1 = accB->f[r];
      if (bias)  { v0 += bias[nc];                     v1 += bias[nc + 16]; }
      if (resid) { v0 += resid[(size_t)gm * 256 + nc]; v1 += resid[(size_t)gm * 256 + nc + 16]; }
      if (relu)  { v0 = fmaxf(v0, 0.f);                v1 = fmaxf(v1, 0.f); }
      C[(size_t)gm * 256 + nc]      = v0;
      C[(size_t)gm * 256 + nc + 16] = v1;
    }
  }
}

// ---------------------------------------------------------------------------
// Pack f32 weight (K x 256) -> f16 fragment-native layout:
//   out[((k>>5)*256 + n)*32 + (k&31)] ; zero-padded to Kpad
// ---------------------------------------------------------------------------
__global__ void k_pack(const float* __restrict__ W, _Float16* __restrict__ out,
                       int K, int Kpad, int Nc)
{
  int idx = blockIdx.x * 256 + threadIdx.x;
  if (idx >= Kpad * Nc) return;
  int k = idx / Nc, n = idx % Nc;
  float v = (k < K) ? W[k * Nc + n] : 0.f;
  out[((size_t)(k >> 5) * 256 + n) * 32 + (k & 31)] = (_Float16)v;
}

// ---------------------------------------------------------------------------
// Gate: g[n][c] = 158 * softmax_c( (gate_in @ gate_W + gate_b) / 5 )
// ---------------------------------------------------------------------------
__global__ __launch_bounds__(64) void k_gate(const float* __restrict__ x,
                                             const float* __restrict__ gW,
                                             const float* __restrict__ gb,
                                             float* __restrict__ g)
{
  int n = blockIdx.x;
  __shared__ float sl[158];
  __shared__ float smax, ssum;
  const float* gin = x + ((size_t)n * 32 + 31) * 221 + 158;
  for (int c = threadIdx.x; c < 158; c += 64) {
    float acc = gb[c];
    for (int j = 0; j < 63; ++j) acc += gin[j] * gW[j * 158 + c];
    sl[c] = acc * 0.2f;
  }
  __syncthreads();
  if (threadIdx.x == 0) {
    float mx = sl[0];
    for (int c = 1; c < 158; ++c) mx = fmaxf(mx, sl[c]);
    float sm = 0.f;
    for (int c = 0; c < 158; ++c) sm += expf(sl[c] - mx);
    smax = mx; ssum = sm;
  }
  __syncthreads();
  for (int c = threadIdx.x; c < 158; c += 64)
    g[(size_t)n * 158 + c] = 158.f * expf(sl[c] - smax) / ssum;
}

// Positional encoding: pe[t][d], t<32, d<256
__global__ void k_pe(float* __restrict__ pe)
{
  int idx = blockIdx.x * 256 + threadIdx.x;
  int t = idx >> 8, d = idx & 255;
  int i2 = d & ~1;
  float div = expf(-logf(10000.f) * (float)i2 / 256.f);
  float ang = (float)t * div;
  pe[idx] = (d & 1) ? cosf(ang) : sinf(ang);
}

// A0[row][k] = src[n,t,k] * g[n][k], padded to K=160
__global__ void k_prep_a0(const float* __restrict__ x, const float* __restrict__ g,
                          float* __restrict__ A0)
{
  size_t idx = (size_t)blockIdx.x * 256 + threadIdx.x;
  int row = (int)(idx / 160), k = (int)(idx % 160);
  int n = row >> 5, t = row & 31;
  float v = 0.f;
  if (k < 158) v = x[((size_t)n * 32 + t) * 221 + k] * g[(size_t)n * 158 + k];
  A0[idx] = v;
}

// h[row][d] += pe[row%32][d]
__global__ __launch_bounds__(256) void k_addpe(float* __restrict__ h,
                                               const float* __restrict__ pe)
{
  int row = blockIdx.x, d = threadIdx.x;
  h[(size_t)row * 256 + d] += pe[((row & 31) << 8) + d];
}

// mean-pool along time
__global__ __launch_bounds__(256) void k_down(const float* __restrict__ h,
                                              float* __restrict__ out,
                                              int factor, int Tout)
{
  int row = blockIdx.x, d = threadIdx.x;
  int n = row / Tout, to = row % Tout;
  float s = 0.f;
  for (int j = 0; j < factor; ++j)
    s += h[((size_t)n * 32 + to * factor + j) * 256 + d];
  out[(size_t)row * 256 + d] = s / (float)factor;
}

// LayerNorm over 256, optional residual add on input
__global__ __launch_bounds__(256) void k_ln(const float* __restrict__ X,
                                            const float* __restrict__ res,
                                            const float* __restrict__ gmm,
                                            const float* __restrict__ bta,
                                            float* __restrict__ Y)
{
  int row = blockIdx.x, d = threadIdx.x;
  __shared__ float red[256];
  float v = X[(size_t)row * 256 + d];
  if (res) v += res[(size_t)row * 256 + d];
  red[d] = v; __syncthreads();
  for (int s = 128; s > 0; s >>= 1) { if (d < s) red[d] += red[d + s]; __syncthreads(); }
  float mean = red[0] / 256.f; __syncthreads();
  float dv = v - mean;
  red[d] = dv * dv; __syncthreads();
  for (int s = 128; s > 0; s >>= 1) { if (d < s) red[d] += red[d + s]; __syncthreads(); }
  float var = red[0] / 256.f;
  Y[(size_t)row * 256 + d] = dv * rsqrtf(var + 1e-5f) * gmm[d] + bta[d];
}

// ---------------------------------------------------------------------------
// Temporal attention (per-stock, nhead=4, dh=64, T<=32) — VALU
// ---------------------------------------------------------------------------
__global__ __launch_bounds__(128) void k_attn_time(
    const float* __restrict__ Q, const float* __restrict__ K,
    const float* __restrict__ V, float* __restrict__ O,
    int T, float invtemp)
{
  int n  = blockIdx.x >> 2;
  int hd = blockIdx.x & 3;
  __shared__ float sQ[32][64], sK[32][64], sV[32][64], sS[32][32];
  int tid = threadIdx.x;
  for (int idx = tid; idx < T * 64; idx += 128) {
    int t = idx >> 6, d = idx & 63;
    size_t base = ((size_t)n * T + t) * 256 + hd * 64 + d;
    sQ[t][d] = Q[base]; sK[t][d] = K[base]; sV[t][d] = V[base];
  }
  __syncthreads();
  for (int idx = tid; idx < T * T; idx += 128) {
    int qi = idx / T, kj = idx % T;
    float acc = 0.f;
    for (int d = 0; d < 64; ++d) acc += sQ[qi][d] * sK[kj][d];
    sS[qi][kj] = acc * invtemp;
  }
  __syncthreads();
  if (tid < T) {
    float mx = -1e30f;
    for (int k2 = 0; k2 < T; ++k2) mx = fmaxf(mx, sS[tid][k2]);
    float sm = 0.f;
    for (int k2 = 0; k2 < T; ++k2) { float e = expf(sS[tid][k2] - mx); sS[tid][k2] = e; sm += e; }
    float inv = 1.f / sm;
    for (int k2 = 0; k2 < T; ++k2) sS[tid][k2] *= inv;
  }
  __syncthreads();
  for (int idx = tid; idx < T * 64; idx += 128) {
    int qi = idx >> 6, d = idx & 63;
    float acc = 0.f;
    for (int kj = 0; kj < T; ++kj) acc += sS[qi][kj] * sV[kj][d];
    O[((size_t)n * T + qi) * 256 + hd * 64 + d] = acc;
  }
}

// ---------------------------------------------------------------------------
// Cross-stock flash attention (nhead=2, dh=128, 1024 keys) — WMMA
// Computes S^T = K@Q^T so P^T registers are directly the A-fragment of P@V.
// grid = (64 qtiles, T*2), block = 32 (one wave)
// ---------------------------------------------------------------------------
__global__ __launch_bounds__(32) void k_attn_stock(
    const float* __restrict__ Q, const float* __restrict__ K,
    const float* __restrict__ V, float* __restrict__ O,
    int T, float invtemp)
{
  int qt   = blockIdx.x;
  int t    = blockIdx.y >> 1;
  int hd   = blockIdx.y & 1;
  int lane = threadIdx.x;
  int lq   = lane & 15;
  int hi   = lane >> 4;

  __shared__ __align__(16) _Float16 sQ[16][128];
  __shared__ __align__(16) _Float16 sK[32][128];
  __shared__ __align__(16) _Float16 sVt[128][32];   // transposed V: [dim][key]

  // stage Q tile (16x128)
  for (int i = lane; i < 512; i += 32) {
    int r = i >> 5, d4 = (i & 31) << 2;
    const float4 f = *(const float4*)&Q[((size_t)(qt * 16 + r) * T + t) * 256 + hd * 128 + d4];
    v4h h = { (_Float16)f.x, (_Float16)f.y, (_Float16)f.z, (_Float16)f.w };
    *(v4h*)&sQ[r][d4] = h;
  }
  __syncthreads();

  // loop-invariant Q B-fragments (lane = query column)
  AF qb[4];
  #pragma unroll
  for (int kc = 0; kc < 4; ++kc) {
    int db = kc * 32 + hi * 16;
    *(v8h*)&qb[kc].h[0] = *(const v8h*)&sQ[lq][db];
    *(v8h*)&qb[kc].h[8] = *(const v8h*)&sQ[lq][db + 8];
  }

  float mrun = -1e30f, lrun = 0.f;
  CF oacc[8];
  #pragma unroll
  for (int f = 0; f < 8; ++f)
    #pragma unroll
    for (int r = 0; r < 8; ++r) oacc[f].f[r] = 0.f;

  const size_t rstride = (size_t)T * 256;

  for (int kt = 0; kt < 32; ++kt) {
    __syncthreads();
    // stage K (32x128 row-major)
    for (int i = lane; i < 1024; i += 32) {
      int r = i >> 5, d4 = (i & 31) << 2;
      const float4 f = *(const float4*)&K[((size_t)(kt * 32 + r) * T + t) * 256 + hd * 128 + d4];
      v4h h = { (_Float16)f.x, (_Float16)f.y, (_Float16)f.z, (_Float16)f.w };
      *(v4h*)&sK[r][d4] = h;
    }
    // stage V transposed (4x4 register transpose)
    #pragma unroll
    for (int it = 0; it < 8; ++it) {
      int gi = lane + 32 * it;
      int k4 = (gi & 7) << 2;
      int d4 = (gi >> 3) << 2;
      const float* vp = &V[((size_t)(kt * 32 + k4) * T + t) * 256 + hd * 128 + d4];
      float4 r0 = *(const float4*)(vp);
      float4 r1 = *(const float4*)(vp + rstride);
      float4 r2 = *(const float4*)(vp + 2 * rstride);
      float4 r3 = *(const float4*)(vp + 3 * rstride);
      v4h c0 = { (_Float16)r0.x, (_Float16)r1.x, (_Float16)r2.x, (_Float16)r3.x };
      v4h c1 = { (_Float16)r0.y, (_Float16)r1.y, (_Float16)r2.y, (_Float16)r3.y };
      v4h c2 = { (_Float16)r0.z, (_Float16)r1.z, (_Float16)r2.z, (_Float16)r3.z };
      v4h c3 = { (_Float16)r0.w, (_Float16)r1.w, (_Float16)r2.w, (_Float16)r3.w };
      *(v4h*)&sVt[d4 + 0][k4] = c0;
      *(v4h*)&sVt[d4 + 1][k4] = c1;
      *(v4h*)&sVt[d4 + 2][k4] = c2;
      *(v4h*)&sVt[d4 + 3][k4] = c3;
    }
    __syncthreads();

    // S^T (32 keys x 16 queries) = K @ Q^T : two C fragments
    CF st0, st1;
    #pragma unroll
    for (int r = 0; r < 8; ++r) { st0.f[r] = 0.f; st1.f[r] = 0.f; }
    #pragma unroll
    for (int kc = 0; kc < 4; ++kc) {
      AF ka0, ka1;
      int kb = kc * 32 + hi * 16;
      *(v8h*)&ka0.h[0] = *(const v8h*)&sK[lq][kb];
      *(v8h*)&ka0.h[8] = *(const v8h*)&sK[lq][kb + 8];
      *(v8h*)&ka1.h[0] = *(const v8h*)&sK[lq + 16][kb];
      *(v8h*)&ka1.h[8] = *(const v8h*)&sK[lq + 16][kb + 8];
      st0.v = WMMA_F32_F16(ka0.v, qb[kc].v, st0.v);
      st1.v = WMMA_F32_F16(ka1.v, qb[kc].v, st1.v);
    }

    // online softmax per query (lane = query column; partner lane via xor 16)
    float p[16];
    float mt = -1e30f;
    #pragma unroll
    for (int r = 0; r < 8; ++r) {
      st0.f[r] *= invtemp; st1.f[r] *= invtemp;
      mt = fmaxf(mt, fmaxf(st0.f[r], st1.f[r]));
    }
    mt = fmaxf(mt, __shfl_xor(mt, 16, 32));
    float mnew = fmaxf(mrun, mt);
    float alpha = expf(mrun - mnew);
    float rs = 0.f;
    #pragma unroll
    for (int r = 0; r < 8; ++r) {
      p[r]     = expf(st0.f[r] - mnew);
      p[8 + r] = expf(st1.f[r] - mnew);
      rs += p[r] + p[8 + r];
    }
    rs += __shfl_xor(rs, 16, 32);
    lrun = lrun * alpha + rs;
    mrun = mnew;

    // redistribute alpha to O-row ownership and rescale accumulators
    float arow[8];
    #pragma unroll
    for (int r = 0; r < 8; ++r) arow[r] = __shfl(alpha, r + 8 * hi, 32);
    #pragma unroll
    for (int f = 0; f < 8; ++f)
      #pragma unroll
      for (int r = 0; r < 8; ++r) oacc[f].f[r] *= arow[r];

    // P A-fragment is exactly the per-lane p[] values
    AF pa;
    #pragma unroll
    for (int e = 0; e < 16; ++e) pa.h[e] = (_Float16)p[e];

    // O += P @ V  (V B-fragments contiguous from transposed LDS)
    #pragma unroll
    for (int f = 0; f < 8; ++f) {
      AF vb;
      *(v8h*)&vb.h[0] = *(const v8h*)&sVt[f * 16 + lq][hi * 8];
      *(v8h*)&vb.h[8] = *(const v8h*)&sVt[f * 16 + lq][hi * 8 + 16];
      oacc[f].v = WMMA_F32_F16(pa.v, vb.v, oacc[f].v);
    }
  }

  // normalize and store
  float lrow[8];
  #pragma unroll
  for (int r = 0; r < 8; ++r) lrow[r] = __shfl(lrun, r + 8 * hi, 32);
  #pragma unroll
  for (int f = 0; f < 8; ++f)
    #pragma unroll
    for (int r = 0; r < 8; ++r) {
      int n = qt * 16 + r + 8 * hi;
      O[((size_t)n * T + t) * 256 + hd * 128 + f * 16 + lq] = oacc[f].f[r] / lrow[r];
    }
}

// ---------------------------------------------------------------------------
// Attention pooling: lam = softmax_t(<hh[n,t], hh[n,-1]>), pooled = sum lam*z
// ---------------------------------------------------------------------------
__global__ __launch_bounds__(256) void k_pool(const float* __restrict__ z,
                                              const float* __restrict__ hh,
                                              float* __restrict__ fin,
                                              int T, int colOff)
{
  int n = blockIdx.x, d = threadIdx.x;
  __shared__ float red[256];
  __shared__ float lam[32];
  float hl = hh[((size_t)n * T + (T - 1)) * 256 + d];
  for (int t = 0; t < T; ++t) {
    red[d] = hl * hh[((size_t)n * T + t) * 256 + d];
    __syncthreads();
    for (int s = 128; s > 0; s >>= 1) { if (d < s) red[d] += red[d + s]; __syncthreads(); }
    if (d == 0) lam[t] = red[0];
    __syncthreads();
  }
  if (d == 0) {
    float mx = -1e30f;
    for (int t = 0; t < T; ++t) mx = fmaxf(mx, lam[t]);
    float sm = 0.f;
    for (int t = 0; t < T; ++t) { float e = expf(lam[t] - mx); lam[t] = e; sm += e; }
    for (int t = 0; t < T; ++t) lam[t] /= sm;
  }
  __syncthreads();
  float acc = 0.f;
  for (int t = 0; t < T; ++t) acc += lam[t] * z[((size_t)n * T + t) * 256 + d];
  fin[(size_t)n * 768 + colOff + d] = acc;
}

// Final: relu(LN(fused)) @ dec_W + dec_b
__global__ __launch_bounds__(256) void k_final(const float* __restrict__ fused,
                                               const float* __restrict__ g,
                                               const float* __restrict__ b,
                                               const float* __restrict__ dW,
                                               const float* __restrict__ db,
                                               float* __restrict__ out)
{
  int n = blockIdx.x, d = threadIdx.x;
  __shared__ float red[256];
  float v = fused[(size_t)n * 256 + d];
  red[d] = v; __syncthreads();
  for (int s = 128; s > 0; s >>= 1) { if (d < s) red[d] += red[d + s]; __syncthreads(); }
  float mean = red[0] / 256.f; __syncthreads();
  float dv = v - mean;
  red[d] = dv * dv; __syncthreads();
  for (int s = 128; s > 0; s >>= 1) { if (d < s) red[d] += red[d + s]; __syncthreads(); }
  float var = red[0] / 256.f; __syncthreads();
  float y = fmaxf(dv * rsqrtf(var + 1e-5f) * g[d] + b[d], 0.f);
  red[d] = y * dW[d]; __syncthreads();
  for (int s = 128; s > 0; s >>= 1) { if (d < s) red[d] += red[d + s]; __syncthreads(); }
  if (d == 0) out[n] = red[0] + db[0];
}

// ---------------------------------------------------------------------------
// Host orchestration
// ---------------------------------------------------------------------------
static void gemm(hipStream_t s, const float* A, int lda, const _Float16* Bp, int Kpad,
                 float* C, int M, const float* bias, const float* resid, int relu)
{
  dim3 grid(4, (unsigned)((M + 127) / 128));
  k_gemm<<<grid, 256, 0, s>>>(A, lda, Bp, Kpad, C, M, bias, resid, relu);
}

extern "C" void kernel_launch(void* const* d_in, const int* in_sizes, int n_in,
                              void* d_out, int out_size, void* d_ws, size_t ws_size,
                              hipStream_t stream)
{
  (void)in_sizes; (void)n_in; (void)out_size; (void)ws_size;
  const float* x        = (const float*)d_in[0];
  const float* gate_W   = (const float*)d_in[1];
  const float* gate_b   = (const float*)d_in[2];
  const float* feat_W   = (const float*)d_in[3];
  const float* feat_b   = (const float*)d_in[4];
  const float* ds_mid_W = (const float*)d_in[5];
  const float* ds_mid_b = (const float*)d_in[6];
  const float* ds_sm_W  = (const float*)d_in[7];
  const float* ds_sm_b  = (const float*)d_in[8];
  const float* tn1g = (const float*)d_in[9];  const float* tn1b = (const float*)d_in[10];
  const float* tWq  = (const float*)d_in[11]; const float* tWk  = (const float*)d_in[12];
  const float* tWv  = (const float*)d_in[13];
  const float* tn2g = (const float*)d_in[14]; const float* tn2b = (const float*)d_in[15];
  const float* tW1  = (const float*)d_in[16]; const float* tb1  = (const float*)d_in[17];
  const float* tW2  = (const float*)d_in[18]; const float* tb2  = (const float*)d_in[19];
  const float* sn1g = (const float*)d_in[20]; const float* sn1b = (const float*)d_in[21];
  const float* sWq  = (const float*)d_in[22]; const float* sWk  = (const float*)d_in[23];
  const float* sWv  = (const float*)d_in[24];
  const float* sn2g = (const float*)d_in[25]; const float* sn2b = (const float*)d_in[26];
  const float* sW1  = (const float*)d_in[27]; const float* sb1  = (const float*)d_in[28];
  const float* sW2  = (const float*)d_in[29]; const float* sb2  = (const float*)d_in[30];
  const float* temp_W = (const float*)d_in[31];
  const float* fus_W  = (const float*)d_in[32]; const float* fus_b  = (const float*)d_in[33];
  const float* fus_g  = (const float*)d_in[34]; const float* fus_bb = (const float*)d_in[35];
  const float* dec_W  = (const float*)d_in[36]; const float* dec_b  = (const float*)d_in[37];

  char* p = (char*)d_ws;
  auto alloc = [&](size_t bytes) -> char* {
    char* r = p; p += (bytes + 255) & ~(size_t)255; return r;
  };

  float* pe = (float*)alloc(32 * 256 * 4);
  float* g  = (float*)alloc(1024 * 158 * 4);
  float* A0 = (float*)alloc((size_t)32768 * 160 * 4);

  _Float16* featP = (_Float16*)alloc(160 * 256 * 2);
  _Float16* Wp[13];
  for (int i = 0; i < 13; ++i) Wp[i] = (_Float16*)alloc(256 * 256 * 2);
  _Float16* dsMidP = Wp[0], *dsSmP = Wp[1];
  _Float16* tWqP = Wp[2], *tWkP = Wp[3], *tWvP = Wp[4], *tW1P = Wp[5], *tW2P = Wp[6];
  _Float16* sWqP = Wp[7], *sWkP = Wp[8], *sWvP = Wp[9], *sW1P = Wp[10], *sW2P = Wp[11];
  _Float16* tempP = Wp[12];
  _Float16* fusP = (_Float16*)alloc(768 * 256 * 2);

  float* Hf = (float*)alloc((size_t)32768 * 256 * 4);
  float* Hm = (float*)alloc((size_t)16384 * 256 * 4);
  float* Hs = (float*)alloc((size_t)8192  * 256 * 4);
  float* S[6];
  for (int i = 0; i < 6; ++i) S[i] = (float*)alloc((size_t)32768 * 256 * 4);
  float* fin   = (float*)alloc(1024 * 768 * 4);
  float* fused = (float*)alloc(1024 * 256 * 4);

  auto pack = [&](const float* W, int K, int Kpad, _Float16* dst) {
    int total = Kpad * 256;
    k_pack<<<(total + 255) / 256, 256, 0, stream>>>(W, dst, K, Kpad, 256);
  };
  pack(feat_W, 158, 160, featP);
  pack(ds_mid_W, 256, 256, dsMidP);  pack(ds_sm_W, 256, 256, dsSmP);
  pack(tWq, 256, 256, tWqP); pack(tWk, 256, 256, tWkP); pack(tWv, 256, 256, tWvP);
  pack(tW1, 256, 256, tW1P); pack(tW2, 256, 256, tW2P);
  pack(sWq, 256, 256, sWqP); pack(sWk, 256, 256, sWkP); pack(sWv, 256, 256, sWvP);
  pack(sW1, 256, 256, sW1P); pack(sW2, 256, 256, sW2P);
  pack(temp_W, 256, 256, tempP);
  pack(fus_W, 768, 768, fusP);

  // stem
  k_pe<<<32, 256, 0, stream>>>(pe);
  k_gate<<<1024, 64, 0, stream>>>(x, gate_W, gate_b, g);
  k_prep_a0<<<(32768 * 160) / 256, 256, 0, stream>>>(x, g, A0);
  gemm(stream, A0, 160, featP, 160, Hf, 32768, feat_b, nullptr, 0);
  k_addpe<<<32768, 256, 0, stream>>>(Hf, pe);
  k_down<<<16384, 256, 0, stream>>>(Hf, S[0], 2, 16);
  gemm(stream, S[0], 256, dsMidP, 256, Hm, 16384, ds_mid_b, nullptr, 0);
  k_down<<<8192, 256, 0, stream>>>(Hf, S[0], 4, 8);
  gemm(stream, S[0], 256, dsSmP, 256, Hs, 8192, ds_sm_b, nullptr, 0);

  const float invtemp_t = 1.f / 8.f;            // sqrt(256/4)
  const float invtemp_s = 1.f / sqrtf(128.f);   // sqrt(256/2)

  auto do_scale = [&](const float* X, int T, int colOff) {
    int M = 1024 * T;
    // temporal attention block
    float *xn = S[0], *q = S[1], *k = S[2], *v = S[3], *o = S[4];
    k_ln<<<M, 256, 0, stream>>>(X, nullptr, tn1g, tn1b, xn);
    gemm(stream, xn, 256, tWqP, 256, q, M, nullptr, nullptr, 0);
    gemm(stream, xn, 256, tWkP, 256, k, M, nullptr, nullptr, 0);
    gemm(stream, xn, 256, tWvP, 256, v, M, nullptr, nullptr, 0);
    k_attn_time<<<4096, 128, 0, stream>>>(q, k, v, o, T, invtemp_t);
    float* xt = S[1];
    k_ln<<<M, 256, 0, stream>>>(xn, o, tn2g, tn2b, xt);
    float* f1 = S[2];
    gemm(stream, xt, 256, tW1P, 256, f1, M, tb1, nullptr, 1);
    float* z1 = S[0];
    gemm(stream, f1, 256, tW2P, 256, z1, M, tb2, xt, 0);

    // cross-stock attention block
    float *xn2 = S[1], *q2 = S[2], *k2 = S[3], *v2 = S[4], *o2 = S[5];
    k_ln<<<M, 256, 0, stream>>>(z1, nullptr, sn1g, sn1b, xn2);
    gemm(stream, xn2, 256, sWqP, 256, q2, M, nullptr, nullptr, 0);
    gemm(stream, xn2, 256, sWkP, 256, k2, M, nullptr, nullptr, 0);
    gemm(stream, xn2, 256, sWvP, 256, v2, M, nullptr, nullptr, 0);
    {
      dim3 grid(64, (unsigned)(T * 2));
      k_attn_stock<<<grid, 32, 0, stream>>>(q2, k2, v2, o2, T, invtemp_s);
    }
    float* xt2 = S[2];
    k_ln<<<M, 256, 0, stream>>>(xn2, o2, sn2g, sn2b, xt2);
    float* f12 = S[3];
    gemm(stream, xt2, 256, sW1P, 256, f12, M, sb1, nullptr, 1);
    float* z2 = S[1];
    gemm(stream, f12, 256, sW2P, 256, z2, M, sb2, xt2, 0);

    // pooling
    float* hh = S[0];
    gemm(stream, z2, 256, tempP, 256, hh, M, nullptr, nullptr, 0);
    k_pool<<<1024, 256, 0, stream>>>(z2, hh, fin, T, colOff);
  };

  do_scale(Hf, 32, 0);
  do_scale(Hm, 16, 256);
  do_scale(Hs, 8, 512);

  // fuse + decode
  gemm(stream, fin, 768, fusP, 768, fused, 1024, fus_b, nullptr, 0);
  k_final<<<1024, 256, 0, stream>>>(fused, fus_g, fus_bb, dec_W, dec_b, (float*)d_out);
}
